// CrossAttention_5815385719305
// MI455X (gfx1250) — matrix operand order repaired and checked by reading, not compiled, not run on hardware
//
#include <hip/hip_runtime.h>
#include <math.h>
#include <stdint.h>

typedef __attribute__((ext_vector_type(16))) _Float16 v16h;
typedef __attribute__((ext_vector_type(8)))  _Float16 v8h;
typedef __attribute__((ext_vector_type(8)))  float    v8f;
typedef __attribute__((ext_vector_type(4)))  float    v4f;

// ---------------------------------------------------------------------------
// WMMA fragment loaders (wave32, V_WMMA_F32_16X16X32_F16)
// A 16x32 f16 layout (ISA 7.12.2): lane m=lane&15, g=lane>>4;
//   halves 0..7  -> K = g*8 + h        (contiguous 8)
//   halves 8..15 -> K = 16 + g*8 + h-8 (contiguous 8)
// B 32x16 f16 layout: lane n=lane&15, g=lane>>4; same K mapping as A
// C/D: VGPR r holds M = r + g*8, N = lane&15
// ---------------------------------------------------------------------------

__device__ __forceinline__ v16h load_a_f16(const _Float16* __restrict__ arow, int k0, int g) {
  const _Float16* p0 = arow + k0 + g * 8;
  const _Float16* p1 = arow + k0 + 16 + g * 8;
  v8h lo = *(const v8h*)p0;
  v8h hi = *(const v8h*)p1;
  v16h a;
#pragma unroll
  for (int i = 0; i < 8; ++i) { a[i] = lo[i]; a[8 + i] = hi[i]; }
  return a;
}

__device__ __forceinline__ v16h load_a_f32(const float* __restrict__ arow, int k0, int g) {
  const float* p0 = arow + k0 + g * 8;
  const float* p1 = arow + k0 + 16 + g * 8;
  v4f f0 = *(const v4f*)p0;
  v4f f1 = *(const v4f*)(p0 + 4);
  v4f f2 = *(const v4f*)p1;
  v4f f3 = *(const v4f*)(p1 + 4);
  v16h a;
#pragma unroll
  for (int i = 0; i < 4; ++i) {
    a[i]      = (_Float16)f0[i];
    a[4 + i]  = (_Float16)f1[i];
    a[8 + i]  = (_Float16)f2[i];
    a[12 + i] = (_Float16)f3[i];
  }
  return a;
}

__device__ __forceinline__ v16h load_b_f16(const _Float16* __restrict__ brow, int k0, int g) {
  const _Float16* p = brow + k0 + g * 8;
  v8h lo = *(const v8h*)p;
  v8h hi = *(const v8h*)(p + 16);
  v16h b;
#pragma unroll
  for (int i = 0; i < 8; ++i) { b[i] = lo[i]; b[8 + i] = hi[i]; }
  return b;
}

__device__ __forceinline__ v8f wmma32(v16h a, v16h b, v8f c) {
  return __builtin_amdgcn_wmma_f32_16x16x32_f16(false, a, false, b, (short)0, c, false, false);
}

// ---------------------------------------------------------------------------
// Projection GEMM: C_f16[M,N] = A_f32[M,K] @ Bt_f16[N,K]^T + bias[N]
// 256 threads = 8 waves (4 in M x 2 in N). Wave tile 32x32 (2x2 WMMA frags),
// block tile 128x64.
// ---------------------------------------------------------------------------
__global__ __launch_bounds__(256)
void k_proj_gemm(const float* __restrict__ A, const _Float16* __restrict__ Bt,
                 const float* __restrict__ bias, _Float16* __restrict__ C,
                 int M, int N, int K) {
  const int lane = threadIdx.x & 31;
  const int wave = threadIdx.x >> 5;
  const int wm = wave & 3, wn = wave >> 2;
  const int g = lane >> 4;
  const int mr = lane & 15;
  const int row0 = blockIdx.x * 128 + wm * 32;
  const int col0 = blockIdx.y * 64 + wn * 32;

  const float*    ar0 = A  + (size_t)(row0 + mr)      * (size_t)K;
  const float*    ar1 = A  + (size_t)(row0 + 16 + mr) * (size_t)K;
  const _Float16* br0 = Bt + (size_t)(col0 + mr)      * (size_t)K;
  const _Float16* br1 = Bt + (size_t)(col0 + 16 + mr) * (size_t)K;

  v8f acc00 = {}, acc01 = {}, acc10 = {}, acc11 = {};
  for (int k0 = 0; k0 < K; k0 += 32) {
    v16h a0 = load_a_f32(ar0, k0, g);
    v16h a1 = load_a_f32(ar1, k0, g);
    v16h b0 = load_b_f16(br0, k0, g);
    v16h b1 = load_b_f16(br1, k0, g);
    acc00 = wmma32(a0, b0, acc00);
    acc01 = wmma32(a0, b1, acc01);
    acc10 = wmma32(a1, b0, acc10);
    acc11 = wmma32(a1, b1, acc11);
  }
  const float bv0 = bias[col0 + mr];
  const float bv1 = bias[col0 + 16 + mr];
#pragma unroll
  for (int ti = 0; ti < 2; ++ti) {
#pragma unroll
    for (int r = 0; r < 8; ++r) {
      const int m = row0 + ti * 16 + g * 8 + r;
      const float c0 = (ti ? acc10[r] : acc00[r]) + bv0;
      const float c1 = (ti ? acc11[r] : acc01[r]) + bv1;
      C[(size_t)m * (size_t)N + col0 + mr]      = (_Float16)c0;
      C[(size_t)m * (size_t)N + col0 + 16 + mr] = (_Float16)c1;
    }
  }
}

// ---------------------------------------------------------------------------
// Batched f16 GEMM, f32 out: C[m,n] = sum_k A[m,k] * Bt[n,k]
// Wave tile 32x32 (2x2 frags), block tile 128x64.
// ---------------------------------------------------------------------------
__global__ __launch_bounds__(256)
void k_gemm_f16_f32(const _Float16* __restrict__ A, const _Float16* __restrict__ Bt,
                    float* __restrict__ C, int M, int N, int K,
                    long long sA, long long sB, long long sC) {
  const int lane = threadIdx.x & 31;
  const int wave = threadIdx.x >> 5;
  const int wm = wave & 3, wn = wave >> 2;
  const int g = lane >> 4;
  const int mr = lane & 15;
  const int row0 = blockIdx.x * 128 + wm * 32;
  const int col0 = blockIdx.y * 64 + wn * 32;

  const _Float16* Ab = A  + (size_t)blockIdx.z * (size_t)sA;
  const _Float16* Bb = Bt + (size_t)blockIdx.z * (size_t)sB;
  float* Cb = C + (size_t)blockIdx.z * (size_t)sC;

  const _Float16* ar0 = Ab + (size_t)(row0 + mr)      * (size_t)K;
  const _Float16* ar1 = Ab + (size_t)(row0 + 16 + mr) * (size_t)K;
  const _Float16* br0 = Bb + (size_t)(col0 + mr)      * (size_t)K;
  const _Float16* br1 = Bb + (size_t)(col0 + 16 + mr) * (size_t)K;

  v8f acc00 = {}, acc01 = {}, acc10 = {}, acc11 = {};
  for (int k0 = 0; k0 < K; k0 += 32) {
    if (k0 + 64 < K) {  // hint next fragments into cache (global_prefetch_b8)
      __builtin_prefetch(ar0 + k0 + 64, 0, 0);
      __builtin_prefetch(br0 + k0 + 64, 0, 0);
    }
    v16h a0 = load_a_f16(ar0, k0, g);
    v16h a1 = load_a_f16(ar1, k0, g);
    v16h b0 = load_b_f16(br0, k0, g);
    v16h b1 = load_b_f16(br1, k0, g);
    acc00 = wmma32(a0, b0, acc00);
    acc01 = wmma32(a0, b1, acc01);
    acc10 = wmma32(a1, b0, acc10);
    acc11 = wmma32(a1, b1, acc11);
  }
#pragma unroll
  for (int ti = 0; ti < 2; ++ti) {
#pragma unroll
    for (int r = 0; r < 8; ++r) {
      const int m = row0 + ti * 16 + g * 8 + r;
      Cb[(size_t)m * (size_t)N + col0 + mr]      = ti ? acc10[r] : acc00[r];
      Cb[(size_t)m * (size_t)N + col0 + 16 + mr] = ti ? acc11[r] : acc01[r];
    }
  }
}

// ---------------------------------------------------------------------------
// Batched GEMM with transposed A access: C[m,n] = sum_k A[k,m] * Bt[n,k]
// A is row-major [K x M]; each wave stages a 32(K)x32(M) A-tile straight into
// its private LDS region with GLOBAL_LOAD_ASYNC_TO_LDS_B128 (ASYNCcnt path,
// no VGPR round-trip). Wave-local s_wait_{dscnt,asynccnt} replace barriers:
// no LDS region is shared across waves.
// Wave tile 32x32 output (2x2 frags), block tile 128x64.
// ---------------------------------------------------------------------------
__global__ __launch_bounds__(256)
void k_gemm_at_f16_f32(const _Float16* __restrict__ A, const _Float16* __restrict__ Bt,
                       float* __restrict__ C, int M, int N, int K,
                       long long sA, long long sB, long long sC) {
  __shared__ _Float16 lds[8][32 * 32];
  const int lane = threadIdx.x & 31;
  const int wave = threadIdx.x >> 5;
  const int wm = wave & 3, wn = wave >> 2;
  const int g = lane >> 4;
  const int mr = lane & 15;
  const int row0 = blockIdx.x * 128 + wm * 32;
  const int col0 = blockIdx.y * 64 + wn * 32;

  const _Float16* Ab = A  + (size_t)blockIdx.z * (size_t)sA;
  const _Float16* Bb = Bt + (size_t)blockIdx.z * (size_t)sB;
  float* Cb = C + (size_t)blockIdx.z * (size_t)sC;

  const _Float16* br0 = Bb + (size_t)(col0 + mr)      * (size_t)K;
  const _Float16* br1 = Bb + (size_t)(col0 + 16 + mr) * (size_t)K;

  // 32-bit LDS offset of this lane's 64B staging slot (low bits of the
  // generic pointer of a __shared__ object are the LDS byte offset).
  const unsigned ldsoff = (unsigned)(uintptr_t)&lds[wave][lane * 32];

  v8f acc00 = {}, acc01 = {}, acc10 = {}, acc11 = {};
  for (int k0 = 0; k0 < K; k0 += 32) {
    // lane l fetches A[k0+l, row0 .. row0+31] (coalesced, 64B/lane) directly
    // into LDS. Wait for the previous iteration's LDS reads first.
    const unsigned long long gaddr =
        (unsigned long long)(Ab + (size_t)(k0 + lane) * (size_t)M + row0);
    asm volatile(
        "s_wait_dscnt 0\n\t"
        "global_load_async_to_lds_b128 %0, %1, off\n\t"
        "global_load_async_to_lds_b128 %0, %1, off offset:16\n\t"
        "global_load_async_to_lds_b128 %0, %1, off offset:32\n\t"
        "global_load_async_to_lds_b128 %0, %1, off offset:48"
        :: "v"(ldsoff), "v"(gaddr) : "memory");
    // overlap the B fragment loads with the async LDS fill
    v16h b0 = load_b_f16(br0, k0, g);
    v16h b1 = load_b_f16(br1, k0, g);
    asm volatile("s_wait_asynccnt 0" ::: "memory");
    v16h a0, a1;
#pragma unroll
    for (int h = 0; h < 16; ++h) {
      const int kk = ((h >> 3) << 4) + g * 8 + (h & 7);   // K within 0..31
      a0[h] = lds[wave][kk * 32 + mr];
      a1[h] = lds[wave][kk * 32 + 16 + mr];
    }
    acc00 = wmma32(a0, b0, acc00);
    acc01 = wmma32(a0, b1, acc01);
    acc10 = wmma32(a1, b0, acc10);
    acc11 = wmma32(a1, b1, acc11);
  }
#pragma unroll
  for (int ti = 0; ti < 2; ++ti) {
#pragma unroll
    for (int r = 0; r < 8; ++r) {
      const int m = row0 + ti * 16 + g * 8 + r;
      Cb[(size_t)m * (size_t)N + col0 + mr]      = ti ? acc10[r] : acc00[r];
      Cb[(size_t)m * (size_t)N + col0 + 16 + mr] = ti ? acc11[r] : acc01[r];
    }
  }
}

// ---------------------------------------------------------------------------
// Row softmax (fp32 in, f16 out). One 256-thread block per row of 2048.
// ---------------------------------------------------------------------------
__global__ __launch_bounds__(256)
void k_row_softmax(const float* __restrict__ S, _Float16* __restrict__ P, int n) {
  __shared__ float red[256];
  const int t = threadIdx.x;
  const float* src = S + (size_t)blockIdx.x * (size_t)n;
  _Float16*    dst = P + (size_t)blockIdx.x * (size_t)n;

  float vals[8];  // n == 2048 -> 8 per thread
  float m = -INFINITY;
#pragma unroll
  for (int i = 0; i < 8; ++i) {
    vals[i] = src[t + i * 256];
    m = fmaxf(m, vals[i]);
  }
  red[t] = m;
  __syncthreads();
  for (int s = 128; s > 0; s >>= 1) {
    if (t < s) red[t] = fmaxf(red[t], red[t + s]);
    __syncthreads();
  }
  const float rowmax = red[0];
  __syncthreads();

  float sum = 0.f;
#pragma unroll
  for (int i = 0; i < 8; ++i) {
    vals[i] = __expf(vals[i] - rowmax);
    sum += vals[i];
  }
  red[t] = sum;
  __syncthreads();
  for (int s = 128; s > 0; s >>= 1) {
    if (t < s) red[t] += red[t + s];
    __syncthreads();
  }
  const float inv = 1.0f / red[0];
#pragma unroll
  for (int i = 0; i < 8; ++i) dst[t + i * 256] = (_Float16)(vals[i] * inv);
}

// ---------------------------------------------------------------------------
// Tiled transpose + fp32->f16 convert: out[b, c, r] = (f16) in[b, r, c]
// block (32,8); grid (cols/32, rows/32, batch)
// ---------------------------------------------------------------------------
__global__ __launch_bounds__(256)
void k_transpose_cvt(const float* __restrict__ in, _Float16* __restrict__ out,
                     int rows, int cols) {
  __shared__ float tile[32][33];
  const float* inb = in + (size_t)blockIdx.z * (size_t)rows * (size_t)cols;
  _Float16*   outb = out + (size_t)blockIdx.z * (size_t)rows * (size_t)cols;
  const int c0 = blockIdx.x * 32;
  const int r0 = blockIdx.y * 32;
#pragma unroll
  for (int j = 0; j < 4; ++j)
    tile[threadIdx.y + j * 8][threadIdx.x] =
        inb[(size_t)(r0 + threadIdx.y + j * 8) * (size_t)cols + c0 + threadIdx.x];
  __syncthreads();
#pragma unroll
  for (int j = 0; j < 4; ++j)
    outb[(size_t)(c0 + threadIdx.y + j * 8) * (size_t)rows + r0 + threadIdx.x] =
        (_Float16)tile[threadIdx.x][threadIdx.y + j * 8];
}

// ---------------------------------------------------------------------------
extern "C" void kernel_launch(void* const* d_in, const int* in_sizes, int n_in,
                              void* d_out, int out_size, void* d_ws, size_t ws_size,
                              hipStream_t stream) {
  (void)in_sizes; (void)n_in; (void)out_size; (void)ws_size;
  constexpr int B = 8, L = 2048, DA = 1024, DB = 1024, H = 512;

  const float* in_a = (const float*)d_in[0];
  const float* in_b = (const float*)d_in[1];
  const float* Wa   = (const float*)d_in[2];
  const float* ba   = (const float*)d_in[3];
  const float* Wb   = (const float*)d_in[4];
  const float* bb   = (const float*)d_in[5];
  float* out = (float*)d_out;

  char* w = (char*)d_ws;
  auto alloc = [&](size_t bytes) -> char* {
    char* p = w;
    w += (bytes + 255) & ~(size_t)255;
    return p;
  };
  _Float16* inTa    = (_Float16*)alloc((size_t)B * DA * L * 2);  // [B][DA][L]
  _Float16* inTb    = (_Float16*)alloc((size_t)B * DB * L * 2);  // [B][DB][L]
  _Float16* WaT     = (_Float16*)alloc((size_t)H * DA * 2);      // [H][DA]
  _Float16* WbT     = (_Float16*)alloc((size_t)H * DB * 2);      // [H][DB]
  _Float16* ma      = (_Float16*)alloc((size_t)B * L * H * 2);   // mapped_a f16
  _Float16* mb      = (_Float16*)alloc((size_t)B * L * H * 2);   // mapped_b f16
  float*    scores  = (float*)alloc((size_t)B * L * L * 4);      // [B][La][Lb]
  float*    scoresT = (float*)alloc((size_t)B * L * L * 4);      // [B][Lb][La]
  _Float16* attn_a  = (_Float16*)alloc((size_t)B * L * L * 2);   // rowsoftmax(scores)
  _Float16* attn_bT = (_Float16*)alloc((size_t)B * L * L * 2);   // rowsoftmax(scoresT)

  const dim3 tb(32, 8, 1);
  // f16 transposed copies of activations and weights (K-contiguous B operands)
  k_transpose_cvt<<<dim3(DA / 32, L / 32, B), tb, 0, stream>>>(in_a, inTa, L, DA);
  k_transpose_cvt<<<dim3(DB / 32, L / 32, B), tb, 0, stream>>>(in_b, inTb, L, DB);
  k_transpose_cvt<<<dim3(H / 32, DA / 32, 1), tb, 0, stream>>>(Wa, WaT, DA, H);
  k_transpose_cvt<<<dim3(H / 32, DB / 32, 1), tb, 0, stream>>>(Wb, WbT, DB, H);

  // projections: mapped = input @ W + b   (fp32 A converted in-register)
  k_proj_gemm<<<dim3((B * L) / 128, H / 64, 1), 256, 0, stream>>>(in_a, WaT, ba, ma, B * L, H, DA);
  k_proj_gemm<<<dim3((B * L) / 128, H / 64, 1), 256, 0, stream>>>(in_b, WbT, bb, mb, B * L, H, DB);

  // scores = ma @ mb^T  and  scoresT = mb @ ma^T  (per batch)
  k_gemm_f16_f32<<<dim3(L / 128, L / 64, B), 256, 0, stream>>>(
      ma, mb, scores, L, L, H, (long long)L * H, (long long)L * H, (long long)L * L);
  k_gemm_f16_f32<<<dim3(L / 128, L / 64, B), 256, 0, stream>>>(
      mb, ma, scoresT, L, L, H, (long long)L * H, (long long)L * H, (long long)L * L);

  // attn_a = softmax(scores, -1); attn_bT = softmax over La == rowsoftmax(scoresT)
  k_row_softmax<<<B * L, 256, 0, stream>>>(scores, attn_a, L);
  k_row_softmax<<<B * L, 256, 0, stream>>>(scoresT, attn_bT, L);

  // output_a[i,d] = sum_j attn_bT[i,j] * in_b[j,d]   (A normal, Bt = inTb)
  k_gemm_f16_f32<<<dim3(L / 128, DB / 64, B), 256, 0, stream>>>(
      attn_bT, inTb, out, L, DB, L,
      (long long)L * L, (long long)DB * L, (long long)L * DB);

  // output_b[j,d] = sum_i attn_a[i,j] * in_a[i,d]    (A async-staged via LDS, Bt = inTa)
  k_gemm_at_f16_f32<<<dim3(L / 128, DA / 64, B), 256, 0, stream>>>(
      attn_a, inTa, out + (size_t)B * L * DB, L, DA, L,
      (long long)L * L, (long long)DA * L, (long long)L * DA);
}
